// SpatialSelfAttention_38431367364729
// MI455X (gfx1250) — matrix operand order repaired and checked by reading, not compiled, not run on hardware
//
#include <hip/hip_runtime.h>
#include <stdint.h>

// ---------- types ----------
typedef __attribute__((ext_vector_type(16))) __bf16 v16bf;
typedef __attribute__((ext_vector_type(8)))  float  v8f;
typedef __attribute__((ext_vector_type(4)))  unsigned u32x4;
typedef __attribute__((ext_vector_type(4)))  unsigned uint32x4;
typedef __attribute__((ext_vector_type(8)))  int      int32x8;
typedef __attribute__((ext_vector_type(4)))  int      int32x4;

#define NB 8
#define CD 256
#define NT 4096           // 64*64 tokens
#define L2E 1.44269504f

union FragU { v16bf v; __bf16 h[16]; unsigned u[8]; u32x4 q[2]; };
union Pack8 { __bf16 h[8]; u32x4 q; };

static __device__ __forceinline__ __bf16 f2bf(float f) { return (__bf16)f; }

static __device__ __forceinline__ v8f zero8() {
  v8f z = {0.f,0.f,0.f,0.f,0.f,0.f,0.f,0.f};
  return z;
}

static __device__ __forceinline__ v8f bfwmma(v16bf a, v16bf b, v8f c) {
  // D = A*B + C, 16x16x32 bf16 -> f32
  return __builtin_amdgcn_wmma_f32_16x16x32_bf16(false, a, false, b,
                                                 (short)0, c, false, false);
}

// A-matrix (16x32 bf16) fragment from a row-major row pointer.
// lane L: M = L%16 (caller passes that row's base), h = L/16.
// layout: vgpr v holds K = 16h+2v, 16h+2v+1 (contiguous, matches B)
static __device__ __forceinline__ v16bf load_a_bf16(const __bf16* rowptr, int h) {
  FragU f;
  const unsigned* p = (const unsigned*)rowptr;
#pragma unroll
  for (int v = 0; v < 4; ++v) f.u[v] = p[8*h + v];
#pragma unroll
  for (int v = 0; v < 4; ++v) f.u[4+v] = p[8*h + 4 + v];
  return f.v;
}

// B-matrix (32x16 bf16) fragment: per-lane 16 contiguous bf16 (lane supplies
// the K-range 16h..16h+15 for its column); two 16B loads.
static __device__ __forceinline__ v16bf load_b_contig(const __bf16* p) {
  FragU f;
  const u32x4* q = (const u32x4*)p;
  f.q[0] = q[0];
  f.q[1] = q[1];
  return f.v;
}

// ---------- Tensor Data Mover: 2D tile -> LDS ----------
// D# per CDNA5 ISA ch.8: group0 = {flags, lds_addr, global_addr, type=2},
// group1 = {data_size=2B, tensor dims/strides, tile dims}. Groups 2/3 zero (<=2D).
static __device__ __forceinline__ unsigned lds_offset(const void* p) {
  // generic LDS pointer = {SHARED_BASE, offset[31:0]} -> low 32 bits are the DS offset
  return (unsigned)(uintptr_t)p;
}

static __device__ __forceinline__ void tdm_load_2d(
    unsigned lds_off, const void* gaddr,
    unsigned tensor_d0, unsigned tensor_d1,
    unsigned tile_d0, unsigned tile_d1, unsigned stride0_elems)
{
  const unsigned long long ga = (unsigned long long)(uintptr_t)gaddr;
  uint32x4 g0;
  g0[0] = 0x1u;                                   // count=1, user descriptor
  g0[1] = lds_off;                                // LDS byte address
  g0[2] = (unsigned)ga;                           // global_addr[31:0]
  g0[3] = (unsigned)(ga >> 32) | 0x80000000u;     // global_addr[56:32] | type=2<<30
  int32x8 g1;
  g1[0] = (int)(1u << 16);                        // workgroup_mask=0, data_size=1 (2 bytes)
  g1[1] = (int)(tensor_d0 << 16);                 // tensor_dim0[15:0]
  g1[2] = (int)((tensor_d0 >> 16) | (tensor_d1 << 16));
  g1[3] = (int)((tensor_d1 >> 16) | (tile_d0 << 16));
  g1[4] = (int)tile_d1;                           // tile_dim1 | tile_dim2=0
  g1[5] = (int)stride0_elems;                     // tensor_dim0_stride[31:0]
  g1[6] = 0;
  g1[7] = 0;
  int32x4 z4 = {0, 0, 0, 0};
#if defined(__clang_major__) && (__clang_major__ >= 23)
  int32x8 z8 = {0, 0, 0, 0, 0, 0, 0, 0};
  __builtin_amdgcn_tensor_load_to_lds(g0, g1, z4, z4, z8, 0);
#else
  __builtin_amdgcn_tensor_load_to_lds(g0, g1, z4, z4, 0);
#endif
}

// ---------- QKV projection: OutT(d,n) = W(d,c) * X(c,n), stored per mode ----------
// mode 0: Q row-major [b][n][256], scaled by 1/16
// mode 1: K row-major [b][n][256]
// mode 2: V d-major  [b][256][n]
__global__ __launch_bounds__(256) void qkv_proj_kernel(
    const float* __restrict__ x, const float* __restrict__ Wq,
    const float* __restrict__ Wk, const float* __restrict__ Wv,
    __bf16* __restrict__ ws)
{
  const int mode = blockIdx.y;
  const float* W = (mode == 0) ? Wq : (mode == 1) ? Wk : Wv;
  __bf16* Qo = ws;
  __bf16* Ko = ws + (size_t)NB * NT * CD;
  __bf16* Vo = ws + 2 * (size_t)NB * NT * CD;

  const int tid  = threadIdx.x;
  const int wave = tid >> 5, lane = tid & 31;
  const int row  = lane & 15, h = lane >> 4;

  const int gw     = blockIdx.x * 8 + wave;   // 0..8191 per mode
  const int nstrip = gw & 63;                 // 64 strips of 64 tokens
  const int dtile  = (gw >> 6) & 15;          // 16 d-tiles
  const int batch  = gw >> 10;                // 8 batches
  const int d0 = dtile * 16;
  const int n0 = nstrip * 64;

  // A fragments: W rows d0..d0+15, K=c in 8 chunks of 32 (f32 pairs -> bf16)
  v16bf A[8];
  {
    const float* wrow = W + (size_t)(d0 + row) * CD;
#pragma unroll
    for (int kc = 0; kc < 8; ++kc) {
      FragU f;
#pragma unroll
      for (int v = 0; v < 4; ++v) {
        const float2 p0 = *(const float2*)(wrow + kc*32 + 16*h + 2*v);
        f.h[2*v]   = f2bf(p0.x);
        f.h[2*v+1] = f2bf(p0.y);
      }
#pragma unroll
      for (int v = 0; v < 4; ++v) {
        const float2 p1 = *(const float2*)(wrow + kc*32 + 16*h + 8 + 2*v);
        f.h[8+2*v]   = f2bf(p1.x);
        f.h[8+2*v+1] = f2bf(p1.y);
      }
      A[kc] = f.v;
    }
  }

  const float* xb = x + (size_t)batch * CD * NT;
#pragma unroll 1
  for (int t = 0; t < 4; ++t) {
    const int nt = n0 + t * 16;
    const int n  = nt + row;
    v8f acc = zero8();
#pragma unroll
    for (int kc = 0; kc < 8; ++kc) {
      // B fragment of X (c x n): lane column n, K = kc*32 + 16h + [0..16)
      FragU bf;
#pragma unroll
      for (int v = 0; v < 8; ++v) {
        const int c = kc*32 + 16*h + 2*v;
        bf.h[2*v]   = f2bf(xb[(size_t)c       * NT + n]);
        bf.h[2*v+1] = f2bf(xb[(size_t)(c + 1) * NT + n]);
      }
      acc = bfwmma(A[kc], bf.v, acc);
    }
    // C layout: component j is element (M = d0 + j + 8h, N = n)
    if (mode == 2) {
#pragma unroll
      for (int j = 0; j < 8; ++j) {
        const int d = d0 + j + 8*h;
        Vo[((size_t)batch * CD + d) * NT + n] = f2bf(acc[j]);
      }
    } else {
      const float scale = (mode == 0) ? 0.0625f : 1.0f;  // 1/sqrt(256) folded into Q
      __bf16* O = (mode == 0) ? Qo : Ko;
      Pack8 pk;
#pragma unroll
      for (int j = 0; j < 8; ++j) pk.h[j] = f2bf(acc[j] * scale);
      *(u32x4*)(O + ((size_t)batch * NT + n) * CD + d0 + 8*h) = pk.q;
    }
  }
}

// ---------- flash attention: TDM double-buffered K/V tiles, 16 q-rows per wave ----------
__global__ __launch_bounds__(256) void flash_attn_kernel(
    const __bf16* __restrict__ ws, float* __restrict__ out)
{
  const __bf16* Q = ws;
  const __bf16* K = ws + (size_t)NB * NT * CD;
  const __bf16* V = ws + 2 * (size_t)NB * NT * CD;  // [b][d][n]

  __shared__ __bf16 Ksh[2][32 * CD];     // 2 x 16KB  (32 keys x 256 d, row-major)
  __shared__ __bf16 Vsh[2][CD * 32];     // 2 x 16KB  (256 d x 32 keys, d-major)
  __shared__ __bf16 Pstage[8 * 16 * 32]; // 1KB per wave

  const int tid  = threadIdx.x;
  const int wave = tid >> 5, lane = tid & 31;
  const int row  = lane & 15, h = lane >> 4;
  const int batch = blockIdx.y;
  const int qr0   = blockIdx.x * 128 + wave * 16;

  const __bf16* Kb = K + (size_t)batch * NT * CD;
  const __bf16* Vb = V + (size_t)batch * CD * NT;

  // Resident Q fragments: 16 rows x 256 d  (8 A-fragments)
  v16bf Qf[8];
  {
    const __bf16* qrow = Q + ((size_t)batch * NT + qr0 + row) * CD;
#pragma unroll
    for (int kc = 0; kc < 8; ++kc) Qf[kc] = load_a_bf16(qrow + kc*32, h);
  }

  v8f Oacc[16];
#pragma unroll
  for (int t = 0; t < 16; ++t) Oacc[t] = zero8();
  float m[8], l[8];
#pragma unroll
  for (int j = 0; j < 8; ++j) { m[j] = -1e30f; l[j] = 0.f; }

  __bf16* Pw = Pstage + wave * (16 * 32);
  const int NCH = NT / 32;               // 128 key chunks

  // prologue: stage chunk 0
  if (wave == 0) {
    tdm_load_2d(lds_offset(&Ksh[0][0]), Kb, CD, NT, CD, 32, CD);
    tdm_load_2d(lds_offset(&Vsh[0][0]), Vb, NT, CD, 32, CD, NT);
  }

#pragma unroll 1
  for (int kb = 0; kb < NCH; ++kb) {
    const int buf = kb & 1;
    const int k0  = kb * 32;

    __syncthreads();                     // everyone done reading back buffer
    if (wave == 0) {
      const int knext = (kb + 1 < NCH) ? (kb + 1) * 32 : 0;  // tail re-load is harmless
      tdm_load_2d(lds_offset(&Ksh[buf ^ 1][0]), Kb + (size_t)knext * CD, CD, NT, CD, 32, CD);
      tdm_load_2d(lds_offset(&Vsh[buf ^ 1][0]), Vb + knext,              NT, CD, 32, CD, NT);
      __builtin_amdgcn_s_wait_tensorcnt(2);  // TDM in-order: current chunk complete
    }
    __syncthreads();                     // front buffer visible to all waves

    // S = Q * K^T for 32 keys (two 16x16 C-tiles), contraction over d=256
    v8f s0 = zero8(), s1 = zero8();
#pragma unroll
    for (int kc = 0; kc < 8; ++kc) {
      const v16bf b0 = load_b_contig(&Ksh[buf][(row)      * CD + kc*32 + 16*h]);
      const v16bf b1 = load_b_contig(&Ksh[buf][(16 + row) * CD + kc*32 + 16*h]);
      s0 = bfwmma(Qf[kc], b0, s0);
      s1 = bfwmma(Qf[kc], b1, s1);
    }

    // Online softmax. C-tile: component j <-> query row j+8h; key = lane%16 (+16 for s1).
    float alpha[8];
#pragma unroll
    for (int j = 0; j < 8; ++j) {
      float t = fmaxf(s0[j], s1[j]);
      t = fmaxf(t, __shfl_xor(t, 1, 32));
      t = fmaxf(t, __shfl_xor(t, 2, 32));
      t = fmaxf(t, __shfl_xor(t, 4, 32));
      t = fmaxf(t, __shfl_xor(t, 8, 32));
      const float mn = fmaxf(m[j], t);
      alpha[j] = __builtin_amdgcn_exp2f((m[j] - mn) * L2E);
      m[j] = mn;
      const float p0 = __builtin_amdgcn_exp2f((s0[j] - mn) * L2E);
      const float p1 = __builtin_amdgcn_exp2f((s1[j] - mn) * L2E);
      s0[j] = p0; s1[j] = p1;
      float rs = p0 + p1;
      rs += __shfl_xor(rs, 1, 32);
      rs += __shfl_xor(rs, 2, 32);
      rs += __shfl_xor(rs, 4, 32);
      rs += __shfl_xor(rs, 8, 32);
      l[j] = l[j] * alpha[j] + rs;
    }

    // Rescale running output
#pragma unroll
    for (int t = 0; t < 16; ++t)
#pragma unroll
      for (int j = 0; j < 8; ++j) Oacc[t][j] *= alpha[j];

    // Stage P (C-layout) into per-wave LDS as row-major 16x32, reload as A-fragment.
    // Same-wave DS ops are in order; no barrier needed.
#pragma unroll
    for (int j = 0; j < 8; ++j) {
      Pw[(j + 8*h) * 32 + row]      = f2bf(s0[j]);
      Pw[(j + 8*h) * 32 + 16 + row] = f2bf(s1[j]);
    }
    const v16bf Pf = load_a_bf16(Pw + row * 32, h);

    // O += P * V over the 16 d-tiles; Vsh is d-major so B-fragments are contiguous.
#pragma unroll
    for (int t = 0; t < 16; ++t) {
      const v16bf vf = load_b_contig(&Vsh[buf][(t*16 + row) * 32 + 16*h]);
      Oacc[t] = bfwmma(Pf, vf, Oacc[t]);
    }
  }

  // Epilogue: O /= l, store to (b, c, h, w) == out[b][d][n]
  float rl[8];
#pragma unroll
  for (int j = 0; j < 8; ++j) rl[j] = __builtin_amdgcn_rcpf(l[j]);
  float* ob = out + (size_t)batch * CD * NT;
#pragma unroll
  for (int t = 0; t < 16; ++t) {
    const int d = t*16 + row;
#pragma unroll
    for (int j = 0; j < 8; ++j)
      ob[(size_t)d * NT + qr0 + j + 8*h] = Oacc[t][j] * rl[j];
  }
}

// ---------- launch ----------
extern "C" void kernel_launch(void* const* d_in, const int* in_sizes, int n_in,
                              void* d_out, int out_size, void* d_ws, size_t ws_size,
                              hipStream_t stream) {
  (void)in_sizes; (void)n_in; (void)out_size; (void)ws_size;
  const float* x  = (const float*)d_in[0];
  const float* Wq = (const float*)d_in[1];
  const float* Wk = (const float*)d_in[2];
  const float* Wv = (const float*)d_in[3];
  __bf16* ws = (__bf16*)d_ws;   // Q | K | V(d-major), bf16: 3*8*4096*256*2 = 48 MB
  float* out = (float*)d_out;

  dim3 pg(1024, 3);             // 8192 waves per mode: 8 batches x 16 d-tiles x 64 strips
  qkv_proj_kernel<<<pg, 256, 0, stream>>>(x, Wq, Wk, Wv, ws);

  dim3 ag(32, 8);               // 32 query blocks (128 rows) x 8 batches
  flash_attn_kernel<<<ag, 256, 0, stream>>>(ws, out);
}
